// AttentionSubLayer_28656021799609
// MI455X (gfx1250) — matrix operand order repaired and checked by reading, not compiled, not run on hardware
//
#include <hip/hip_runtime.h>

#define B_   2
#define T_   2048
#define D_   2048
#define H_   16
#define KVH_ 4
#define HD_  128

typedef __attribute__((ext_vector_type(16))) __bf16 bf16x16;
typedef __attribute__((ext_vector_type(8)))  float  f32x8;

union FragBF { bf16x16 v; uint4 u[2]; };

__device__ __forceinline__ f32x8 wmma_bf16(bf16x16 a, bf16x16 b, f32x8 c) {
  return __builtin_amdgcn_wmma_f32_16x16x32_bf16(false, a, false, b, (short)0, c, false, false);
}

// ---------------------------------------------------------------------------
// time-shift mix (x + mix*(shift(x)-x)) and fp32 -> bf16 convert
__global__ __launch_bounds__(256) void k_mix_bf16(const float* __restrict__ x,
                                                  const float* __restrict__ mix,
                                                  __bf16* __restrict__ o) {
  long idx = (long)blockIdx.x * 256 + threadIdx.x;   // over (B*T*D)/4
  long row = idx >> 9;                               // D/4 = 512 chunks per row
  int  c4  = ((int)idx & 511) << 2;
  int  t   = (int)(row & (T_ - 1));
  const float4 xv = *(const float4*)&x[row * D_ + c4];
  float4 r = xv;
  if (mix != nullptr) {
    const float4 mv = *(const float4*)&mix[c4];
    float4 xp = make_float4(0.f, 0.f, 0.f, 0.f);
    if (t > 0) xp = *(const float4*)&x[(row - 1) * D_ + c4];
    r.x = xv.x + mv.x * (xp.x - xv.x);
    r.y = xv.y + mv.y * (xp.y - xv.y);
    r.z = xv.z + mv.z * (xp.z - xv.z);
    r.w = xv.w + mv.w * (xp.w - xv.w);
  }
  __bf16* op = o + row * D_ + c4;
  op[0] = (__bf16)r.x; op[1] = (__bf16)r.y; op[2] = (__bf16)r.z; op[3] = (__bf16)r.w;
}

// ---------------------------------------------------------------------------
// weight convert + transpose: Wt[n][k] = (bf16) W[k][n], K fixed at 2048
__global__ __launch_bounds__(256) void k_wconv_t(const float* __restrict__ W,
                                                 __bf16* __restrict__ Wt, int N) {
  long idx = (long)blockIdx.x * 256 + threadIdx.x;   // exactly K*N threads
  long k = idx / N;
  long n = idx % N;
  Wt[n * 2048 + k] = (__bf16)W[idx];
}

// ---------------------------------------------------------------------------
// WMMA GEMM: C[M=grid.x*256][N=grid.y*128] = A[M][2048] @ Bt[N][2048]^T
// 8 waves; each wave owns 32 rows x 128 cols (2 A-frags x 8 B-frags = 16 WMMA/K-step).
// Register-staged prefetch of the next K-tile hides global latency.
// MODE 0: plain bf16, natural [row][Ntotal] layout (gate projection)
// MODE 1: RMS-norm epilogue, bf16 head-major [(b*heads+h)*T+t][128] (V)
// MODE 2: RMS-norm + RoPE epilogue, bf16 head-major (Q, K)
// MODE 3: plain fp32, natural layout (output projection)
template <int MODE>
__global__ __launch_bounds__(256) void k_gemm_wmma(const __bf16* __restrict__ A,
                                                   const __bf16* __restrict__ Bt,
                                                   void* __restrict__ Out,
                                                   int Ntotal, int heads) {
  __shared__ __align__(16) __bf16 lA[256][40];
  __shared__ __align__(16) __bf16 lB[128][40];
  const int tid  = threadIdx.x;
  const int lane = tid & 31, wave = tid >> 5;
  const int n16  = lane & 15, hi = lane >> 4;
  const long rowBase = (long)blockIdx.x * 256;
  const long nBase   = (long)blockIdx.y * 128;

  uint4 pa[4], pb[2];
  int ra[4], ca[4], rb[2], cb2[2];
#pragma unroll
  for (int i = 0; i < 4; ++i) {
    int c = tid + i * 256;            // 1024 chunks of 16B for A (256x32)
    ra[i] = c >> 2;
    ca[i] = (c & 3) << 3;
  }
#pragma unroll
  for (int i = 0; i < 2; ++i) {
    int c = tid + i * 256;            // 512 chunks of 16B for B (128x32)
    rb[i] = c >> 2;
    cb2[i] = (c & 3) << 3;
  }

  // prologue: load K-tile 0 into registers, store to LDS
#pragma unroll
  for (int i = 0; i < 4; ++i) pa[i] = *(const uint4*)&A[(rowBase + ra[i]) * 2048 + ca[i]];
#pragma unroll
  for (int i = 0; i < 2; ++i) pb[i] = *(const uint4*)&Bt[(nBase + rb[i]) * 2048 + cb2[i]];
#pragma unroll
  for (int i = 0; i < 4; ++i) *(uint4*)&lA[ra[i]][ca[i]] = pa[i];
#pragma unroll
  for (int i = 0; i < 2; ++i) *(uint4*)&lB[rb[i]][cb2[i]] = pb[i];
  __syncthreads();

  f32x8 acc[2][8];
#pragma unroll
  for (int m = 0; m < 2; ++m)
#pragma unroll
    for (int j = 0; j < 8; ++j)
#pragma unroll
      for (int e = 0; e < 8; ++e) acc[m][j][e] = 0.f;

  for (int kk = 0; kk < 2048; kk += 32) {
    const bool more = (kk + 32) < 2048;
    if (more) {  // issue next tile's global loads early; compute hides the latency
#pragma unroll
      for (int i = 0; i < 4; ++i) pa[i] = *(const uint4*)&A[(rowBase + ra[i]) * 2048 + kk + 32 + ca[i]];
#pragma unroll
      for (int i = 0; i < 2; ++i) pb[i] = *(const uint4*)&Bt[(nBase + rb[i]) * 2048 + kk + 32 + cb2[i]];
    }
    FragBF fa[2];
#pragma unroll
    for (int m = 0; m < 2; ++m) {
      const int am = wave * 32 + m * 16 + n16;
      fa[m].u[0] = *(const uint4*)&lA[am][hi * 8];
      fa[m].u[1] = *(const uint4*)&lA[am][16 + hi * 8];
    }
#pragma unroll
    for (int j = 0; j < 8; ++j) {
      FragBF fb;
      fb.u[0] = *(const uint4*)&lB[j * 16 + n16][hi * 8];
      fb.u[1] = *(const uint4*)&lB[j * 16 + n16][16 + hi * 8];
      acc[0][j] = wmma_bf16(fa[0].v, fb.v, acc[0][j]);
      acc[1][j] = wmma_bf16(fa[1].v, fb.v, acc[1][j]);
    }
    if (more) {
      __syncthreads();  // all waves done reading current tile
#pragma unroll
      for (int i = 0; i < 4; ++i) *(uint4*)&lA[ra[i]][ca[i]] = pa[i];
#pragma unroll
      for (int i = 0; i < 2; ++i) *(uint4*)&lB[rb[i]][cb2[i]] = pb[i];
      __syncthreads();
    }
  }

  if (MODE == 0 || MODE == 3) {
#pragma unroll
    for (int m = 0; m < 2; ++m)
#pragma unroll
      for (int r = 0; r < 8; ++r) {
        const long row = rowBase + wave * 32 + m * 16 + r + hi * 8;
#pragma unroll
        for (int j = 0; j < 8; ++j) {
          const long col = nBase + j * 16 + n16;
          if (MODE == 0) ((__bf16*)Out)[row * Ntotal + col] = (__bf16)acc[m][j][r];
          else           ((float*)Out)[row * Ntotal + col]  = acc[m][j][r];
        }
      }
    return;
  }

  // RMS-norm (+RoPE) epilogue; N-tile == one head (128) so reductions are wave-local
  const int h = blockIdx.y;
#pragma unroll
  for (int m = 0; m < 2; ++m)
#pragma unroll
    for (int r = 0; r < 8; ++r) {
      float ss = 0.f;
#pragma unroll
      for (int j = 0; j < 8; ++j) ss += acc[m][j][r] * acc[m][j][r];
      ss += __shfl_xor(ss, 1, 16);
      ss += __shfl_xor(ss, 2, 16);
      ss += __shfl_xor(ss, 4, 16);
      ss += __shfl_xor(ss, 8, 16);
      const float rstd = rsqrtf(ss * (1.f / 128.f) + 1e-8f);
      float v[8];
#pragma unroll
      for (int j = 0; j < 8; ++j) v[j] = acc[m][j][r] * rstd;
      const long rowG = rowBase + wave * 32 + m * 16 + r + hi * 8;
      const int  t    = (int)(rowG & (T_ - 1));
      const long b    = rowG >> 11;
      float wv[8];
      if (MODE == 2) {
#pragma unroll
        for (int j = 0; j < 8; ++j) {
          const int c  = j * 16 + n16;
          const int fi = c & 63;
          const float ang = (float)t * __expf((float)fi * (-9.210340371976184f / 64.f));
          float sn, cs;
          __sincosf(ang, &sn, &cs);
          const float rot = (j < 4) ? -v[j + 4] : v[j - 4];
          wv[j] = v[j] * cs + rot * sn;
        }
      } else {
#pragma unroll
        for (int j = 0; j < 8; ++j) wv[j] = v[j];
      }
      __bf16* ob = (__bf16*)Out;
      const long base = ((b * heads + h) * (long)T_ + t) * HD_;
#pragma unroll
      for (int j = 0; j < 8; ++j) ob[base + j * 16 + n16] = (__bf16)wv[j];
    }
}

// ---------------------------------------------------------------------------
// causal flash attention + fused 1/l, L2-norm, gating
// grid = (T/128, B*H); block 256 = 8 waves, wave owns 16 q-rows.
// K/V tiles are register-staged one iteration ahead of the compute.
__global__ __launch_bounds__(256) void k_attn(const __bf16* __restrict__ qb,
                                              const __bf16* __restrict__ kb,
                                              const __bf16* __restrict__ vb,
                                              const __bf16* __restrict__ gb,
                                              __bf16* __restrict__ yb) {
  __shared__ __align__(16) __bf16 lK[64][136];
  __shared__ __align__(16) __bf16 lVt[128][72];
  __shared__ __align__(16) __bf16 lP[8][16][72];

  const int tid  = threadIdx.x;
  const int lane = tid & 31, wave = tid >> 5;
  const int n16  = lane & 15, hi = lane >> 4;
  const int bh = blockIdx.y;
  const int b  = bh >> 4, h = bh & 15;
  const int kvh = h >> 2;                       // GQA: 4 q-heads per kv-head
  const int q0  = blockIdx.x * 128;

  const __bf16* Q = qb + (long)(b * H_ + h) * T_ * HD_;
  const __bf16* K = kb + (long)(b * KVH_ + kvh) * T_ * HD_;
  const __bf16* V = vb + (long)(b * KVH_ + kvh) * T_ * HD_;

  // Q fragments stay resident in registers for the whole key loop
  bf16x16 qf[4];
  const int myq = q0 + wave * 16 + n16;
#pragma unroll
  for (int ks = 0; ks < 4; ++ks) {
    FragBF f;
    const __bf16* p = Q + (long)myq * HD_ + ks * 32 + hi * 8;
    f.u[0] = *(const uint4*)p;
    f.u[1] = *(const uint4*)(p + 16);
    qf[ks] = f.v;
  }

  // per-thread K/V tile staging registers (64 keys x 128 hd per tile)
  uint4 pk[4], pv[4];
  int rr[4], cc[4];
#pragma unroll
  for (int i = 0; i < 4; ++i) {
    int c = tid + i * 256;
    rr[i] = c >> 4;
    cc[i] = (c & 15) << 3;
  }
#pragma unroll
  for (int i = 0; i < 4; ++i) {
    pk[i] = *(const uint4*)&K[(long)rr[i] * HD_ + cc[i]];
    pv[i] = *(const uint4*)&V[(long)rr[i] * HD_ + cc[i]];
  }

  f32x8 O[8];
#pragma unroll
  for (int j = 0; j < 8; ++j)
#pragma unroll
    for (int e = 0; e < 8; ++e) O[j][e] = 0.f;
  float mrow[8], lrow[8];
#pragma unroll
  for (int r = 0; r < 8; ++r) { mrow[r] = -3.0e38f; lrow[r] = 0.f; }

  const int ntiles = (q0 >> 6) + 2;              // causal: keys up to q0+127
  for (int kt = 0; kt < ntiles; ++kt) {
    __syncthreads();                              // previous tile's readers done
#pragma unroll
    for (int i = 0; i < 4; ++i) {
      *(uint4*)&lK[rr[i]][cc[i]] = pk[i];
      const __bf16* e8 = (const __bf16*)&pv[i];
#pragma unroll
      for (int e = 0; e < 8; ++e) lVt[cc[i] + e][rr[i]] = e8[e];  // transpose V
    }
    __syncthreads();
    if (kt + 1 < ntiles) {                        // stage next tile during compute
      const int kn = (kt + 1) * 64;
#pragma unroll
      for (int i = 0; i < 4; ++i) {
        pk[i] = *(const uint4*)&K[(long)(kn + rr[i]) * HD_ + cc[i]];
        pv[i] = *(const uint4*)&V[(long)(kn + rr[i]) * HD_ + cc[i]];
      }
    }
    const int k0 = kt * 64;

    // S = Q @ K^T (K tile used directly as B operand: n=key, k=hd)
    f32x8 s[4];
#pragma unroll
    for (int jn = 0; jn < 4; ++jn) {
#pragma unroll
      for (int e = 0; e < 8; ++e) s[jn][e] = 0.f;
#pragma unroll
      for (int ks = 0; ks < 4; ++ks) {
        FragBF fb;
        fb.u[0] = *(const uint4*)&lK[jn * 16 + n16][ks * 32 + hi * 8];
        fb.u[1] = *(const uint4*)&lK[jn * 16 + n16][ks * 32 + 16 + hi * 8];
        s[jn] = wmma_bf16(qf[ks], fb.v, s[jn]);
      }
    }
    const float sc = 0.08838834764831845f;       // 1/sqrt(128)
#pragma unroll
    for (int jn = 0; jn < 4; ++jn)
#pragma unroll
      for (int r = 0; r < 8; ++r) {
        const int key = k0 + jn * 16 + n16;
        const int qi  = q0 + wave * 16 + r + hi * 8;
        const float val = s[jn][r] * sc;
        s[jn][r] = (key <= qi) ? val : -3.0e38f;
      }

    // online softmax, wave-local per row
#pragma unroll
    for (int r = 0; r < 8; ++r) {
      float mx = s[0][r];
#pragma unroll
      for (int jn = 1; jn < 4; ++jn) mx = fmaxf(mx, s[jn][r]);
      mx = fmaxf(mx, __shfl_xor(mx, 1, 16));
      mx = fmaxf(mx, __shfl_xor(mx, 2, 16));
      mx = fmaxf(mx, __shfl_xor(mx, 4, 16));
      mx = fmaxf(mx, __shfl_xor(mx, 8, 16));
      const float mnew = fmaxf(mrow[r], mx);
      const float fsc  = __expf(mrow[r] - mnew);
      float ps = 0.f;
#pragma unroll
      for (int jn = 0; jn < 4; ++jn) {
        const float p = __expf(s[jn][r] - mnew);
        s[jn][r] = p;
        ps += p;
      }
      ps += __shfl_xor(ps, 1, 16);
      ps += __shfl_xor(ps, 2, 16);
      ps += __shfl_xor(ps, 4, 16);
      ps += __shfl_xor(ps, 8, 16);
      lrow[r] = lrow[r] * fsc + ps;
      mrow[r] = mnew;
#pragma unroll
      for (int j = 0; j < 8; ++j) O[j][r] *= fsc;
    }

    // P through wave-private LDS (C-layout -> A-layout transpose)
#pragma unroll
    for (int jn = 0; jn < 4; ++jn)
#pragma unroll
      for (int r = 0; r < 8; ++r)
        lP[wave][r + hi * 8][jn * 16 + n16] = (__bf16)s[jn][r];
    bf16x16 pf[2];
#pragma unroll
    for (int ks = 0; ks < 2; ++ks) {
      FragBF f;
      const __bf16* p = &lP[wave][n16][ks * 32 + hi * 8];
      f.u[0] = *(const uint4*)p;
      f.u[1] = *(const uint4*)(p + 16);
      pf[ks] = f.v;
    }
    // O += P @ V (Vt: n=hd, k=key)
#pragma unroll
    for (int j = 0; j < 8; ++j)
#pragma unroll
      for (int ks = 0; ks < 2; ++ks) {
        FragBF fv;
        fv.u[0] = *(const uint4*)&lVt[j * 16 + n16][ks * 32 + hi * 8];
        fv.u[1] = *(const uint4*)&lVt[j * 16 + n16][ks * 32 + 16 + hi * 8];
        O[j] = wmma_bf16(pf[ks], fv.v, O[j]);
      }
  }

  // finalize: 1/l, L2-norm over HD, gate, write bf16 natural layout
#pragma unroll
  for (int r = 0; r < 8; ++r) {
    const float inv = 1.f / lrow[r];
    float ss = 0.f;
#pragma unroll
    for (int j = 0; j < 8; ++j) {
      O[j][r] *= inv;
      ss += O[j][r] * O[j][r];
    }
    ss += __shfl_xor(ss, 1, 16);
    ss += __shfl_xor(ss, 2, 16);
    ss += __shfl_xor(ss, 4, 16);
    ss += __shfl_xor(ss, 8, 16);
    const float l2 = 1.f / fmaxf(sqrtf(ss), 1e-12f);
    const int  t    = q0 + wave * 16 + r + hi * 8;
    const long base = ((long)(b * T_ + t)) * D_ + h * HD_;
#pragma unroll
    for (int j = 0; j < 8; ++j) {
      const long idx = base + j * 16 + n16;
      const float gv = (float)gb[idx];
      yb[idx] = (__bf16)(O[j][r] * l2 * gv);
    }
  }
}

// ---------------------------------------------------------------------------
// final RMS-norm / sqrt(48); one block per row of 2048
__global__ __launch_bounds__(256) void k_final_norm(const float* __restrict__ o,
                                                    float* __restrict__ out) {
  const int row = blockIdx.x;
  const int tid = threadIdx.x;
  __shared__ float red[8];
  float v[8];
  float ss = 0.f;
  const long base = (long)row * D_;
#pragma unroll
  for (int i = 0; i < 8; ++i) {
    v[i] = o[base + tid + i * 256];
    ss += v[i] * v[i];
  }
  ss += __shfl_xor(ss, 1);
  ss += __shfl_xor(ss, 2);
  ss += __shfl_xor(ss, 4);
  ss += __shfl_xor(ss, 8);
  ss += __shfl_xor(ss, 16);
  if ((tid & 31) == 0) red[tid >> 5] = ss;
  __syncthreads();
  float tot = 0.f;
#pragma unroll
  for (int i = 0; i < 8; ++i) tot += red[i];
  const float scale = rsqrtf(tot * (1.f / 2048.f) + 1e-8f) * 0.14433756729740643f;
#pragma unroll
  for (int i = 0; i < 8; ++i) out[base + tid + i * 256] = v[i] * scale;
}

// ---------------------------------------------------------------------------
extern "C" void kernel_launch(void* const* d_in, const int* in_sizes, int n_in,
                              void* d_out, int out_size, void* d_ws, size_t ws_size,
                              hipStream_t stream) {
  (void)in_sizes; (void)n_in; (void)out_size; (void)ws_size;
  const float* xq = (const float*)d_in[0];
  const float* xk = (const float*)d_in[1];
  const float* xv = (const float*)d_in[2];
  const float* Wq = (const float*)d_in[3];
  const float* Wk = (const float*)d_in[4];
  const float* Wv = (const float*)d_in[5];
  const float* Wg = (const float*)d_in[6];
  const float* Wo = (const float*)d_in[7];
  const float* mk = (const float*)d_in[8];
  const float* mv = (const float*)d_in[9];
  float* out = (float*)d_out;

  char*  w   = (char*)d_ws;
  size_t off = 0;
  auto alloc = [&](size_t bytes) -> void* {
    void* p = w + off;
    off += (bytes + 255) & ~(size_t)255;
    return p;
  };
  const size_t MR = (size_t)B_ * T_;  // 4096 rows
  __bf16* xqb = (__bf16*)alloc(MR * D_ * 2);
  __bf16* xkb = (__bf16*)alloc(MR * D_ * 2);
  __bf16* xvb = (__bf16*)alloc(MR * D_ * 2);
  __bf16* Wqt = (__bf16*)alloc((size_t)2048 * 2048 * 2);
  __bf16* Wkt = (__bf16*)alloc((size_t)512 * 2048 * 2);
  __bf16* Wvt = (__bf16*)alloc((size_t)512 * 2048 * 2);
  __bf16* Wgt = (__bf16*)alloc((size_t)2048 * 2048 * 2);
  __bf16* Wot = (__bf16*)alloc((size_t)2048 * 2048 * 2);
  __bf16* qbuf = (__bf16*)alloc((size_t)B_ * H_ * T_ * HD_ * 2);
  __bf16* kbuf = (__bf16*)alloc((size_t)B_ * KVH_ * T_ * HD_ * 2);
  __bf16* vbuf = (__bf16*)alloc((size_t)B_ * KVH_ * T_ * HD_ * 2);
  __bf16* gbuf = (__bf16*)alloc(MR * D_ * 2);
  __bf16* ybuf = (__bf16*)alloc(MR * D_ * 2);
  float*  obuf = (float*)alloc(MR * D_ * 4);

  // stage 1: mix + convert
  k_mix_bf16<<<8192, 256, 0, stream>>>(xq, nullptr, xqb);
  k_mix_bf16<<<8192, 256, 0, stream>>>(xk, mk, xkb);
  k_mix_bf16<<<8192, 256, 0, stream>>>(xv, mv, xvb);
  k_wconv_t<<<16384, 256, 0, stream>>>(Wq, Wqt, 2048);
  k_wconv_t<<<4096, 256, 0, stream>>>(Wk, Wkt, 512);
  k_wconv_t<<<4096, 256, 0, stream>>>(Wv, Wvt, 512);
  k_wconv_t<<<16384, 256, 0, stream>>>(Wg, Wgt, 2048);
  k_wconv_t<<<16384, 256, 0, stream>>>(Wo, Wot, 2048);

  // stage 2: projections with fused norm/rope epilogues (M-tile 256)
  k_gemm_wmma<2><<<dim3(16, 16), 256, 0, stream>>>(xqb, Wqt, qbuf, 2048, H_);
  k_gemm_wmma<2><<<dim3(16, 4), 256, 0, stream>>>(xkb, Wkt, kbuf, 512, KVH_);
  k_gemm_wmma<1><<<dim3(16, 4), 256, 0, stream>>>(xvb, Wvt, vbuf, 512, KVH_);
  k_gemm_wmma<0><<<dim3(16, 16), 256, 0, stream>>>(xqb, Wgt, gbuf, 2048, H_);

  // stage 3: flash attention + L2-norm + gate
  k_attn<<<dim3(16, 32), 256, 0, stream>>>(qbuf, kbuf, vbuf, gbuf, ybuf);

  // stage 4: output projection + final norm
  k_gemm_wmma<3><<<dim3(16, 16), 256, 0, stream>>>(ybuf, Wot, obuf, 2048, H_);
  k_final_norm<<<4096, 256, 0, stream>>>(obuf, out);
}
